// AttentionLayer_16028817949357
// MI455X (gfx1250) — matrix operand order repaired and checked by reading, not compiled, not run on hardware
//
#include <hip/hip_runtime.h>
#include <hip/hip_bf16.h>

// ---------------------------------------------------------------------------
// Problem constants: B=2, L=2048, HID=1024, H=16, DH=64
// ---------------------------------------------------------------------------
#define BB    2
#define LL    2048
#define HIDD  1024
#define HH    16
#define DHH   64
#define MM    (BB * LL)          // 4096 rows for all projections

typedef __bf16 bf16_t;
typedef __attribute__((ext_vector_type(16))) __bf16 v16bf;
typedef __attribute__((ext_vector_type(8)))  float  v8f;

// fp32 -> bf16 round-to-nearest-even
static __device__ __forceinline__ bf16_t f2b(float f) {
    union { float f; unsigned int u; } v; v.f = f;
    unsigned int r = v.u + 0x7FFFu + ((v.u >> 16) & 1u);
    unsigned short h = (unsigned short)(r >> 16);
    bf16_t b;
    __builtin_memcpy(&b, &h, 2);
    return b;
}

// ---------------------------------------------------------------------------
// CDNA5 async global->LDS copy (ASYNCcnt-tracked, no VGPR staging).
// LDS address operand = low 32 bits of the flat pointer (LDS aperture maps by
// truncation, ISA 10.2).  16 bytes per lane per instruction.
// ---------------------------------------------------------------------------
static __device__ __forceinline__ void async_copy_b128(const void* gptr, void* lptr) {
    unsigned lds = (unsigned)(size_t)lptr;
    asm volatile("global_load_async_to_lds_b128 %0, %1, off"
                 :
                 : "v"(lds), "v"(gptr)
                 : "memory");
}
static __device__ __forceinline__ void async_wait0() {
    asm volatile("s_wait_asynccnt 0x0" ::: "memory");
}

// ---------------------------------------------------------------------------
// WMMA fragment loaders (CDNA5 ISA 7.12.2, wave32)
// ---------------------------------------------------------------------------
static __device__ __forceinline__ v16bf load_frag_a(const bf16_t* p, int stride, int lane) {
    const int r  = lane & 15;
    const int kh = (lane >> 4) * 16;
    v16bf f;
#pragma unroll
    for (int j = 0; j < 8; ++j) {
        const int k = kh + j * 2;
        f[2 * j]     = p[r * stride + k];
        f[2 * j + 1] = p[r * stride + k + 1];
    }
    return f;
}
// B(32x16): col N = lane&15; K half by lane half. Element (k,n) at p[n*stride+k].
static __device__ __forceinline__ v16bf load_frag_b_nk(const bf16_t* p, int stride, int lane) {
    const int n  = lane & 15;
    const int kh = (lane >> 4) * 16;
    v16bf f;
#pragma unroll
    for (int j = 0; j < 16; ++j) f[j] = p[n * stride + kh + j];
    return f;
}
// Same layout, element (k,n) at p[k*stride+n].
static __device__ __forceinline__ v16bf load_frag_b_kn(const bf16_t* p, int stride, int lane) {
    const int n  = lane & 15;
    const int kh = (lane >> 4) * 16;
    v16bf f;
#pragma unroll
    for (int j = 0; j < 16; ++j) f[j] = p[(kh + j) * stride + n];
    return f;
}

static __device__ __forceinline__ v8f wmma_bf16(v16bf a, v16bf b, v8f c) {
    return __builtin_amdgcn_wmma_f32_16x16x32_bf16(false, a, false, b, (short)0, c, false, false);
}

// ---------------------------------------------------------------------------
// Bulk fp32 -> bf16 convert (8 elems / thread, b128-friendly)
// ---------------------------------------------------------------------------
__global__ __launch_bounds__(256) void cvt_f32_bf16_kernel(
    const float* __restrict__ src, bf16_t* __restrict__ dst, int n)
{
    const int i = (blockIdx.x * 256 + threadIdx.x) * 8;
    if (i < n) {
#pragma unroll
        for (int j = 0; j < 8; ++j) dst[i + j] = f2b(src[i + j]);
    }
}

// ---------------------------------------------------------------------------
// GEMM: C[M,N] = A[M,K] @ W[N,K]^T + bias, all-bf16 operands, fp32 accumulate.
// Block tile 128x128, K-step 64.  8 waves: 2(M) x 4(N); wave tile 64x32
// (4x2 WMMA tiles) -> 16 WMMAs per staging round.  Staging via async LDS copies.
// ---------------------------------------------------------------------------
#define GBM 128
#define GBN 128
#define GBK 64
#define GPADB 8
#define GLDST (GBK + GPADB)      // 72 bf16 = 144 B (16B aligned)

template <bool OUT_F32>
__global__ __launch_bounds__(256) void gemm_bf16_wmma_kernel(
    const bf16_t* __restrict__ A, const bf16_t* __restrict__ W,
    const float* __restrict__ bias, void* __restrict__ Cv,
    int M, int N, int K)
{
    __shared__ bf16_t sA[GBM][GLDST];
    __shared__ bf16_t sB[GBN][GLDST];

    const int tid  = threadIdx.x;
    const int lane = tid & 31;
    const int wave = tid >> 5;
    const int wm   = wave & 1;           // 0..1  (64 rows each)
    const int wn   = wave >> 1;          // 0..3  (32 cols each)
    const int m0   = blockIdx.y * GBM;
    const int n0   = blockIdx.x * GBN;

    v8f acc[4][2] = {};

    const int sr = tid >> 1;             // staging row 0..127
    const int sc = (tid & 1) * 32;       // staging col base (32 bf16 = 64 B)

    for (int k0 = 0; k0 < K; k0 += GBK) {
        const bf16_t* ga = A + (size_t)(m0 + sr) * K + k0 + sc;
        const bf16_t* gw = W + (size_t)(n0 + sr) * K + k0 + sc;
#pragma unroll
        for (int i = 0; i < 4; ++i) {
            async_copy_b128(ga + i * 8, &sA[sr][sc + i * 8]);
            async_copy_b128(gw + i * 8, &sB[sr][sc + i * 8]);
        }
        async_wait0();
        __syncthreads();

#pragma unroll
        for (int ks = 0; ks < 2; ++ks) {
            v16bf af[4], bf[2];
#pragma unroll
            for (int mi = 0; mi < 4; ++mi)
                af[mi] = load_frag_a(&sA[wm * 64 + mi * 16][ks * 32], GLDST, lane);
#pragma unroll
            for (int ni = 0; ni < 2; ++ni)
                bf[ni] = load_frag_b_nk(&sB[wn * 32 + ni * 16][ks * 32], GLDST, lane);
#pragma unroll
            for (int mi = 0; mi < 4; ++mi)
#pragma unroll
                for (int ni = 0; ni < 2; ++ni)
                    acc[mi][ni] = wmma_bf16(af[mi], bf[ni], acc[mi][ni]);
        }
        __syncthreads();
    }

    const int half = lane >> 4, ln = lane & 15;
#pragma unroll
    for (int mi = 0; mi < 4; ++mi)
#pragma unroll
        for (int ni = 0; ni < 2; ++ni)
#pragma unroll
            for (int i = 0; i < 8; ++i) {
                const int row = m0 + wm * 64 + mi * 16 + half * 8 + i;
                const int col = n0 + wn * 32 + ni * 16 + ln;
                const float v = acc[mi][ni][i] + bias[col];
                if (OUT_F32) ((float*)Cv)[(size_t)row * N + col] = v;
                else         ((bf16_t*)Cv)[(size_t)row * N + col] = f2b(v);
            }
}

// ---------------------------------------------------------------------------
// Flash attention (bf16 WMMA, fp32 accumulate, online softmax).
// Grid (L/128, B*H); 8 waves; wave owns 16 q rows.  KV in 64-wide tiles.
// ---------------------------------------------------------------------------
#define FQT 128
#define FJT 64
#define FPAD 8
#define FST (DHH + FPAD)   // 72

__global__ __launch_bounds__(256) void attn_wmma_kernel(
    const bf16_t* __restrict__ Qp, const bf16_t* __restrict__ Kp,
    const bf16_t* __restrict__ Vp, const int* __restrict__ mask,
    bf16_t* __restrict__ Ob)
{
    __shared__ bf16_t sQ[FQT][FST];
    __shared__ bf16_t sK[FJT][FST];
    __shared__ bf16_t sV[FJT][FST];
    __shared__ float  sMaskAdd[FJT];
    __shared__ bf16_t sP[8][16][FST];

    const int tid  = threadIdx.x;
    const int lane = tid & 31;
    const int wave = tid >> 5;
    const int b    = blockIdx.y / HH;
    const int h    = blockIdx.y % HH;
    const int q0   = blockIdx.x * FQT;
    const size_t headoff = (size_t)h * DHH;
    const int half = lane >> 4, ln = lane & 15;

    // stage Q tile 128x64 via async copies (4 x b128 per thread)
    {
        const int r = tid >> 1;
        const int c = (tid & 1) * 32;
        const bf16_t* Qb = Qp + (size_t)(b * LL + q0 + r) * HIDD + headoff + c;
#pragma unroll
        for (int i = 0; i < 4; ++i)
            async_copy_b128(Qb + i * 8, &sQ[r][c + i * 8]);
    }
    async_wait0();
    __syncthreads();

    v16bf qfrag[2];
    qfrag[0] = load_frag_a(&sQ[wave * 16][0],  FST, lane);
    qfrag[1] = load_frag_a(&sQ[wave * 16][32], FST, lane);

    v8f   o_acc[4] = {};
    float m_run[8], l_run[8];
#pragma unroll
    for (int i = 0; i < 8; ++i) { m_run[i] = -1e30f; l_run[i] = 0.0f; }

    const float scale = 0.125f;   // DH^-0.5

    for (int j0 = 0; j0 < LL; j0 += FJT) {
        __syncthreads();   // previous sK/sV fully consumed
        {
            const int r = tid >> 2;
            const int c = (tid & 3) * 16;
            const bf16_t* Kb = Kp + (size_t)(b * LL + j0 + r) * HIDD + headoff + c;
            const bf16_t* Vb = Vp + (size_t)(b * LL + j0 + r) * HIDD + headoff + c;
            async_copy_b128(Kb,     &sK[r][c]);
            async_copy_b128(Kb + 8, &sK[r][c + 8]);
            async_copy_b128(Vb,     &sV[r][c]);
            async_copy_b128(Vb + 8, &sV[r][c + 8]);
        }
        if (tid < FJT)
            sMaskAdd[tid] = mask[(size_t)b * LL + j0 + tid] ? 0.0f : -10000.0f;
        async_wait0();
        __syncthreads();

        // S = Q @ K^T
        v8f s_acc[4] = {};
#pragma unroll
        for (int ni = 0; ni < 4; ++ni) {
            v16bf kf0 = load_frag_b_nk(&sK[ni * 16][0],  FST, lane);
            v16bf kf1 = load_frag_b_nk(&sK[ni * 16][32], FST, lane);
            s_acc[ni] = wmma_bf16(qfrag[0], kf0, s_acc[ni]);
            s_acc[ni] = wmma_bf16(qfrag[1], kf1, s_acc[ni]);
        }

        // scale + additive mask
#pragma unroll
        for (int ni = 0; ni < 4; ++ni) {
            const float madd = sMaskAdd[ni * 16 + ln];
#pragma unroll
            for (int i = 0; i < 8; ++i)
                s_acc[ni][i] = s_acc[ni][i] * scale + madd;
        }

        // online softmax; row r = half*8+i lives on the 16 lanes of this half
#pragma unroll
        for (int i = 0; i < 8; ++i) {
            float mx = s_acc[0][i];
#pragma unroll
            for (int ni = 1; ni < 4; ++ni) mx = fmaxf(mx, s_acc[ni][i]);
#pragma unroll
            for (int off = 1; off < 16; off <<= 1)
                mx = fmaxf(mx, __shfl_xor(mx, off, 32));
            const float mnew  = fmaxf(m_run[i], mx);
            const float alpha = __expf(m_run[i] - mnew);
            m_run[i] = mnew;

            float rs = 0.0f;
#pragma unroll
            for (int ni = 0; ni < 4; ++ni) {
                const float p = __expf(s_acc[ni][i] - mnew);
                rs += p;
                sP[wave][half * 8 + i][ni * 16 + ln] = f2b(p);
            }
#pragma unroll
            for (int off = 1; off < 16; off <<= 1)
                rs += __shfl_xor(rs, off, 32);
            l_run[i] = l_run[i] * alpha + rs;
#pragma unroll
            for (int ni = 0; ni < 4; ++ni) o_acc[ni][i] *= alpha;
        }

        // O += P @ V
        v16bf pf0 = load_frag_a(&sP[wave][0][0],  FST, lane);
        v16bf pf1 = load_frag_a(&sP[wave][0][32], FST, lane);
#pragma unroll
        for (int ni = 0; ni < 4; ++ni) {
            v16bf vf0 = load_frag_b_kn(&sV[0][ni * 16],  FST, lane);
            v16bf vf1 = load_frag_b_kn(&sV[32][ni * 16], FST, lane);
            o_acc[ni] = wmma_bf16(pf0, vf0, o_acc[ni]);
            o_acc[ni] = wmma_bf16(pf1, vf1, o_acc[ni]);
        }
    }

    // normalize + store O (bf16) in [b, l, h*DH + d]
#pragma unroll
    for (int ni = 0; ni < 4; ++ni)
#pragma unroll
        for (int i = 0; i < 8; ++i) {
            const int r = wave * 16 + half * 8 + i;
            const float v = o_acc[ni][i] / l_run[i];
            Ob[(size_t)(b * LL + q0 + r) * HIDD + headoff + ni * 16 + ln] = f2b(v);
        }
}

// ---------------------------------------------------------------------------
// Launcher
// ---------------------------------------------------------------------------
extern "C" void kernel_launch(void* const* d_in, const int* in_sizes, int n_in,
                              void* d_out, int out_size, void* d_ws, size_t ws_size,
                              hipStream_t stream) {
    (void)in_sizes; (void)n_in; (void)out_size; (void)ws_size;

    const float* q    = (const float*)d_in[0];
    const float* k    = (const float*)d_in[1];
    const float* v    = (const float*)d_in[2];
    const int*   mask = (const int*)  d_in[3];
    const float* Wq   = (const float*)d_in[4];
    const float* bq   = (const float*)d_in[5];
    const float* Wk   = (const float*)d_in[6];
    const float* bk   = (const float*)d_in[7];
    const float* Wv   = (const float*)d_in[8];
    const float* bv   = (const float*)d_in[9];
    const float* Wo   = (const float*)d_in[10];
    const float* bo   = (const float*)d_in[11];
    float* out = (float*)d_out;

    const size_t ae = (size_t)MM * HIDD;       // activation elems (4M)
    const size_t we = (size_t)HIDD * HIDD;     // weight elems (1M)

    bf16_t* qb  = (bf16_t*)d_ws;
    bf16_t* kb  = qb  + ae;
    bf16_t* vb  = kb  + ae;
    bf16_t* Wqb = vb  + ae;
    bf16_t* Wkb = Wqb + we;
    bf16_t* Wvb = Wkb + we;
    bf16_t* Wob = Wvb + we;
    bf16_t* Qp  = Wob + we;
    bf16_t* Kp  = Qp  + ae;
    bf16_t* Vp  = Kp  + ae;
    bf16_t* Ob  = Vp  + ae;

    // 1) one-shot fp32 -> bf16 conversion of activations and weights
    const int acvt = (int)(ae / 8 / 256);      // 2048 blocks
    const int wcvt = (int)(we / 8 / 256);      // 512 blocks
    cvt_f32_bf16_kernel<<<acvt, 256, 0, stream>>>(q,  qb,  (int)ae);
    cvt_f32_bf16_kernel<<<acvt, 256, 0, stream>>>(k,  kb,  (int)ae);
    cvt_f32_bf16_kernel<<<acvt, 256, 0, stream>>>(v,  vb,  (int)ae);
    cvt_f32_bf16_kernel<<<wcvt, 256, 0, stream>>>(Wq, Wqb, (int)we);
    cvt_f32_bf16_kernel<<<wcvt, 256, 0, stream>>>(Wk, Wkb, (int)we);
    cvt_f32_bf16_kernel<<<wcvt, 256, 0, stream>>>(Wv, Wvb, (int)we);
    cvt_f32_bf16_kernel<<<wcvt, 256, 0, stream>>>(Wo, Wob, (int)we);

    // 2) Q/K/V projections (bf16 out)
    dim3 gblk(256), ggrid(HIDD / GBN, MM / GBM);   // (8, 32)
    gemm_bf16_wmma_kernel<false><<<ggrid, gblk, 0, stream>>>(qb, Wqb, bq, Qp, MM, HIDD, HIDD);
    gemm_bf16_wmma_kernel<false><<<ggrid, gblk, 0, stream>>>(kb, Wkb, bk, Kp, MM, HIDD, HIDD);
    gemm_bf16_wmma_kernel<false><<<ggrid, gblk, 0, stream>>>(vb, Wvb, bv, Vp, MM, HIDD, HIDD);

    // 3) attention
    dim3 ablk(256), agrid(LL / FQT, BB * HH);      // (16, 32)
    attn_wmma_kernel<<<agrid, ablk, 0, stream>>>(Qp, Kp, Vp, mask, Ob);

    // 4) output projection (fp32 out)
    gemm_bf16_wmma_kernel<true><<<ggrid, gblk, 0, stream>>>(Ob, Wob, bo, out, MM, HIDD, HIDD);
}
